// Three_Bodies_RNN_64630667870356
// MI455X (gfx1250) — matrix operand order repaired and checked by reading, not compiled, not run on hardware
//
#include <hip/hip_runtime.h>

#define BB 32
#define SS 4096
#define II 64
#define HH 128
#define OO 64
#define NWG 8

typedef __attribute__((ext_vector_type(16))) __bf16 v16bf;
typedef __attribute__((ext_vector_type(8)))  __bf16 v8bf;
typedef __attribute__((ext_vector_type(8)))  float  v8f;
typedef __attribute__((ext_vector_type(4)))  float  v4f;

static __device__ inline v8f wmma_bf16(v16bf a, v16bf b, v8f c) {
  // D = A(16x32 bf16) x B(32x16 bf16) + C(16x16 f32)
  return __builtin_amdgcn_wmma_f32_16x16x32_bf16(false, a, false, b, (short)0, c,
                                                 false, false);
}

// ---------------------------------------------------------------------------
// Prepass: swizzle T3 (fp32 -> bf16) into WMMA 16-bit A-fragment layout,
// swizzle W_in likewise, copy hidden -> hbuf[0], reset barrier counters.
// A-fragment element map (this code's contiguous k order, not the ISA table order; 16-bit A 16x32):
//   lane l: M = l%16, half = l/16
//   element e: K = half*16 + e
// ---------------------------------------------------------------------------
__global__ __launch_bounds__(256) void prep_kernel(
    const float* __restrict__ T3, const float* __restrict__ W_in,
    const float* __restrict__ hidden,
    __bf16* __restrict__ Afrag, __bf16* __restrict__ Wfrag,
    float* __restrict__ hbuf, unsigned* __restrict__ bar)
{
  const unsigned idx = blockIdx.x * 256u + threadIdx.x;
  if (idx == 0u) { bar[0] = 0u; bar[1] = 0u; }

  if (idx < 2097152u) {                       // T3: 1024 row-tiles x 4 kb x 32 x 16
    const unsigned e    = idx & 15u;
    const unsigned lane = (idx >> 4) & 31u;
    const unsigned kb   = (idx >> 9) & 3u;
    const unsigned R    = idx >> 11;          // row/16, row = i*128 + j
    const unsigned half = lane >> 4, m = lane & 15u;
    const unsigned kl   = half * 16u + e;
    const unsigned row  = R * 16u + m;
    Afrag[idx] = (__bf16)T3[(size_t)row * 128u + kb * 32u + kl];
  } else if (idx < 2097152u + 8192u) {        // W_in: 8 row-tiles x 2 kb x 32 x 16
    const unsigned t    = idx - 2097152u;
    const unsigned e    = t & 15u;
    const unsigned lane = (t >> 4) & 31u;
    const unsigned kb2  = (t >> 9) & 1u;
    const unsigned R    = t >> 10;
    const unsigned half = lane >> 4, m = lane & 15u;
    const unsigned kl   = half * 16u + e;
    Wfrag[t] = (__bf16)W_in[(R * 16u + m) * 64u + kb2 * 32u + kl];
  } else if (idx < 2097152u + 8192u + 4096u) {
    const unsigned t = idx - 2097152u - 8192u;
    hbuf[t] = hidden[t];                      // h ping-pong buffer 0
  }
}

// ---------------------------------------------------------------------------
// Recurrent scan: 8 workgroups (one 16-row i-chunk each), 256 threads = 8 waves.
// Per step: P = A_bf16 x h^T via WMMA, weighted j-reduction in VALU,
// input projection folded in as 4 WMMAs (wave 0), tanh, global barrier.
// ---------------------------------------------------------------------------
__global__ __launch_bounds__(256, 1) void rnn_scan_kernel(
    const float* __restrict__ u,
    const __bf16* __restrict__ Afrag,
    const __bf16* __restrict__ Wfrag,
    const float* __restrict__ b_in,
    const float* __restrict__ bias_t,
    float* __restrict__ hbuf,
    float* __restrict__ dout,
    unsigned* __restrict__ bar)
{
  __shared__ __align__(32) __bf16 hBT[BB * HH];   // h as bf16, [b][k]
  __shared__ __align__(32) __bf16 uBT[BB * II];   // u[:,s,:] as bf16, [b][k]
  __shared__ __align__(16) float  iwLDS[BB][16];  // input-proj slice [b][i_local]

  const int tid  = threadIdx.x;
  const int wv   = tid >> 5;
  const int lane = tid & 31;
  const int n    = lane & 15;
  const int half = lane >> 4;
  const int base_i = blockIdx.x * 16;

  // Persistent W_in A-fragments for this workgroup (K = 64 -> two kb blocks)
  v16bf wfr[2];
#pragma unroll
  for (int kb2 = 0; kb2 < 2; ++kb2)
    wfr[kb2] = *(const v16bf*)(Wfrag +
               (size_t)(((blockIdx.x * 2 + kb2) * 32 + lane) * 16));

  float bsum[2];
#pragma unroll
  for (int il = 0; il < 2; ++il) {
    const int i = base_i + 2 * wv + il;
    bsum[il] = b_in[i] + bias_t[i];
  }

  float* hlast = dout + (size_t)BB * SS * OO;
  float* rout  = hlast + (size_t)BB * HH;
  unsigned* cnt = bar;
  unsigned* gen = bar + 1;

  for (int s = 0; s < SS; ++s) {
    const float* hin  = hbuf + (s & 1) * (BB * HH);
    float*       hout = hbuf + ((s + 1) & 1) * (BB * HH);

    // ---- cooperative staging: h and u[:,s,:] into LDS as bf16 ----
    {
      const int b = tid >> 3;
      const int c = (tid & 7) * 16;
      const float* hp = hin + b * HH + c;
      v4f f0 = *(const v4f*)(hp);
      v4f f1 = *(const v4f*)(hp + 4);
      v4f f2 = *(const v4f*)(hp + 8);
      v4f f3 = *(const v4f*)(hp + 12);
      v8bf o0 = { (__bf16)f0[0], (__bf16)f0[1], (__bf16)f0[2], (__bf16)f0[3],
                  (__bf16)f1[0], (__bf16)f1[1], (__bf16)f1[2], (__bf16)f1[3] };
      v8bf o1 = { (__bf16)f2[0], (__bf16)f2[1], (__bf16)f2[2], (__bf16)f2[3],
                  (__bf16)f3[0], (__bf16)f3[1], (__bf16)f3[2], (__bf16)f3[3] };
      *(v8bf*)(hBT + b * HH + c)     = o0;
      *(v8bf*)(hBT + b * HH + c + 8) = o1;

      const int c2 = (tid & 7) * 8;
      const float* up = u + ((size_t)b * SS + s) * II + c2;
      v4f g0 = *(const v4f*)(up);
      v4f g1 = *(const v4f*)(up + 4);
      v8bf p0 = { (__bf16)g0[0], (__bf16)g0[1], (__bf16)g0[2], (__bf16)g0[3],
                  (__bf16)g1[0], (__bf16)g1[1], (__bf16)g1[2], (__bf16)g1[3] };
      *(v8bf*)(uBT + b * II + c2) = p0;
    }
    __syncthreads();

    // ---- B-matrix fragments for h (32x16 bf16 per kb/nt; lane = column b) ----
    v16bf hfr[4][2];
#pragma unroll
    for (int kb = 0; kb < 4; ++kb)
#pragma unroll
      for (int nt = 0; nt < 2; ++nt)
        hfr[kb][nt] = *(const v16bf*)(hBT + (nt * 16 + n) * HH +
                                      kb * 32 + half * 16);

    // ---- input projection iw[b, i-chunk] (wave 0 only; EXEC stays all-1) ----
    if (wv == 0) {
#pragma unroll
      for (int nt = 0; nt < 2; ++nt) {
        v8f acc = {0.f,0.f,0.f,0.f,0.f,0.f,0.f,0.f};
#pragma unroll
        for (int kb2 = 0; kb2 < 2; ++kb2) {
          v16bf ufr = *(const v16bf*)(uBT + (nt * 16 + n) * II +
                                      kb2 * 32 + half * 16);
          acc = wmma_bf16(wfr[kb2], ufr, acc);
        }
#pragma unroll
        for (int r = 0; r < 8; ++r)
          iwLDS[nt * 16 + n][r + 8 * half] = acc[r];
      }
    }

    // ---- main bilinear contraction: P tiles + weighted j-reduction ----
    float hw00 = 0.f, hw01 = 0.f, hw10 = 0.f, hw11 = 0.f;
#pragma unroll
    for (int il = 0; il < 2; ++il) {
      const int i_local = 2 * wv + il;
      const size_t Rbase = (size_t)(base_i + i_local) * 8;
#pragma unroll 2
      for (int jt = 0; jt < 8; ++jt) {
        const __bf16* ap = Afrag + ((Rbase + jt) * 4 * 32 + lane) * 16;
        __builtin_prefetch(ap + 2048, 0, 1);   // next row-tile of T3 stream
        v8f acc0 = {0.f,0.f,0.f,0.f,0.f,0.f,0.f,0.f};
        v8f acc1 = {0.f,0.f,0.f,0.f,0.f,0.f,0.f,0.f};
#pragma unroll
        for (int kb = 0; kb < 4; ++kb) {
          v16bf afr = *(const v16bf*)(ap + kb * (32 * 16));
          acc0 = wmma_bf16(afr, hfr[kb][0], acc0);
          acc1 = wmma_bf16(afr, hfr[kb][1], acc1);
        }
        // hw[b,i] += sum_m h[b, jt*16+m] * P[m, b]; lane holds m = r + 8*half
        const __bf16* wp = hBT + n * HH + jt * 16 + half * 8;
        v8bf w0 = *(const v8bf*)wp;
        v8bf w1 = *(const v8bf*)(wp + 16 * HH);
        float s0 = 0.f, s1 = 0.f;
#pragma unroll
        for (int r = 0; r < 8; ++r) {
          s0 += (float)w0[r] * acc0[r];
          s1 += (float)w1[r] * acc1[r];
        }
        if (il == 0) { hw00 += s0; hw01 += s1; }
        else         { hw10 += s0; hw11 += s1; }
      }
    }
    __syncthreads();   // iwLDS ready for all waves

    // ---- finalize: combine lane halves, tanh, write h_new / r_out ----
    float hv[4] = { hw00, hw01, hw10, hw11 };
#pragma unroll
    for (int il = 0; il < 2; ++il) {
      const int i_local = 2 * wv + il;
      const int i = base_i + i_local;
#pragma unroll
      for (int nt = 0; nt < 2; ++nt) {
        float v = hv[il * 2 + nt];
        v += __shfl_xor(v, 16, 32);            // merge m-halves (same column b)
        const int b = nt * 16 + n;
        const float hn = tanhf(v + iwLDS[b][i_local] + bsum[il]);
        if (lane < 16) {
          hout[b * HH + i] = hn;
          rout[((size_t)b * SS + s) * HH + i] = hn;
          if (s == SS - 1) hlast[b * HH + i] = hn;
        }
      }
    }

    // ---- device-wide sense barrier (8 resident WGs) ----
    __syncthreads();
    if (tid == 0) {
      __threadfence();
      const unsigned prev = __hip_atomic_fetch_add(cnt, 1u, __ATOMIC_ACQ_REL,
                                                   __HIP_MEMORY_SCOPE_AGENT);
      if (prev == NWG - 1) {
        __hip_atomic_store(cnt, 0u, __ATOMIC_RELAXED, __HIP_MEMORY_SCOPE_AGENT);
        __hip_atomic_fetch_add(gen, 1u, __ATOMIC_ACQ_REL,
                               __HIP_MEMORY_SCOPE_AGENT);
      } else {
        while (__hip_atomic_load(gen, __ATOMIC_ACQUIRE,
                                 __HIP_MEMORY_SCOPE_AGENT) < (unsigned)(s + 1)) {
          __builtin_amdgcn_s_sleep(2);
        }
      }
    }
    __syncthreads();
    __threadfence();
  }
}

// ---------------------------------------------------------------------------
// Output projection: out[b,s,o] = r[b,s,:] . W_out[o,:] + b_out[o]
// Bandwidth-bound (96 MB @ 23.3 TB/s); plain fp32.
// ---------------------------------------------------------------------------
__global__ __launch_bounds__(256) void outproj_kernel(
    const float* __restrict__ rout, const float* __restrict__ W_out,
    const float* __restrict__ b_out, float* __restrict__ out)
{
  const unsigned idx = blockIdx.x * 256u + threadIdx.x;   // b*S*O space
  const unsigned o  = idx & 63u;
  const unsigned bs = idx >> 6;
  const float* r = rout + (size_t)bs * HH;
  const float* w = W_out + o * HH;
  float acc = b_out[o];
#pragma unroll 8
  for (int h = 0; h < HH; h += 4) {
    v4f rv = *(const v4f*)(r + h);
    v4f wv = *(const v4f*)(w + h);
    acc += rv[0]*wv[0] + rv[1]*wv[1] + rv[2]*wv[2] + rv[3]*wv[3];
  }
  out[idx] = acc;
}

extern "C" void kernel_launch(void* const* d_in, const int* in_sizes, int n_in,
                              void* d_out, int out_size, void* d_ws, size_t ws_size,
                              hipStream_t stream) {
  (void)in_sizes; (void)n_in; (void)out_size; (void)ws_size;
  const float* u      = (const float*)d_in[0];
  const float* hidden = (const float*)d_in[1];
  const float* W_in   = (const float*)d_in[2];
  const float* b_in   = (const float*)d_in[3];
  const float* T3     = (const float*)d_in[4];
  const float* bias_t = (const float*)d_in[5];
  const float* W_out  = (const float*)d_in[6];
  const float* b_out  = (const float*)d_in[7];
  float* out = (float*)d_out;

  char* ws = (char*)d_ws;
  __bf16*   Afrag = (__bf16*)ws;                               // 4,194,304 B
  __bf16*   Wfrag = (__bf16*)(ws + 4194304);                   //    16,384 B
  float*    hbuf  = (float*)(ws + 4194304 + 16384);            //    32,768 B
  unsigned* bar   = (unsigned*)(ws + 4194304 + 16384 + 32768); //         8 B

  // (2097152 + 8192 + 4096) / 256 = 8240 blocks
  prep_kernel<<<8240, 256, 0, stream>>>(T3, W_in, hidden, Afrag, Wfrag, hbuf, bar);
  rnn_scan_kernel<<<NWG, 256, 0, stream>>>(u, Afrag, Wfrag, b_in, bias_t,
                                           hbuf, out, bar);
  const float* rout = out + (size_t)BB * SS * OO + (size_t)BB * HH;
  outproj_kernel<<<(BB * SS * OO) / 256, 256, 0, stream>>>(rout, W_out, b_out, out);
}
